// MultiHeadSelfAttention_75187697483848
// MI455X (gfx1250) — matrix operand order repaired and checked by reading, not compiled, not run on hardware
//
#include <hip/hip_runtime.h>

typedef __attribute__((ext_vector_type(16))) __bf16          v16bf;
typedef __attribute__((ext_vector_type(8)))  float           v8f;
typedef __attribute__((ext_vector_type(8)))  unsigned short  u16x8;
typedef __attribute__((ext_vector_type(4)))  unsigned int    u32x4;
typedef __attribute__((ext_vector_type(8)))  int             i32x8;
typedef __attribute__((ext_vector_type(4)))  int             i32x4;

union FragU { v16bf bf; u16x8 u[2]; };

#define HAVE_TDM (__has_builtin(__builtin_amdgcn_tensor_load_to_lds) && \
                  __has_builtin(__builtin_amdgcn_s_wait_tensorcnt))

__device__ __forceinline__ unsigned short f2bf(float f) {
  unsigned int u = __float_as_uint(f);
  u += 0x7fffu + ((u >> 16) & 1u);       // round-to-nearest-even
  return (unsigned short)(u >> 16);
}

#if HAVE_TDM
// 2D tile DMA: global (row-major, stride_elems) -> LDS (packed rows).
// Tensor dims == tile dims (tiles are always fully in-bounds here).
// D# packing per CDNA5 ISA.
// 6-arg builtin form (clang-23 / therock-10.0): (g0, g1, g2, g3, g4, cpol).
__device__ __forceinline__ void tdm_load_2d(unsigned lds_addr, const void* gptr,
                                            unsigned tile_d0, unsigned tile_d1,
                                            unsigned stride_elems) {
  unsigned long long ga = (unsigned long long)(size_t)gptr;
  u32x4 g0;
  g0[0] = 1u;                                   // count=1, user descriptor
  g0[1] = lds_addr;                             // lds_addr[31:0]
  g0[2] = (unsigned)ga;                         // global_addr[31:0] (bits 95:64)
  g0[3] = (unsigned)(ga >> 32) | (2u << 30);    // global_addr[56:32] | type=2
  i32x8 g1;
  g1[0] = 0x00010000;                           // wg_mask=0, data_size=1 (2B)
  g1[1] = (int)((tile_d0 & 0xFFFFu) << 16);     // tensor_dim0[15:0]
  g1[2] = (int)((tile_d0 >> 16) | ((tile_d1 & 0xFFFFu) << 16));  // td0 hi | td1 lo
  g1[3] = (int)((tile_d1 >> 16) | (tile_d0 << 16));              // td1 hi | tile_dim0
  g1[4] = (int)(tile_d1 & 0xFFFFu);             // tile_dim1 | tile_dim2=0
  g1[5] = (int)stride_elems;                    // tensor_dim0_stride[31:0]
  g1[6] = 0;                                    // stride hi | dim1_stride lo
  g1[7] = 0;
  i32x4 gz4 = {0, 0, 0, 0};                     // 2D: groups 2/3 unused
  i32x8 gz8 = {0, 0, 0, 0, 0, 0, 0, 0};
  __builtin_amdgcn_tensor_load_to_lds(g0, g1, gz4, gz4, gz8, 0);
}
#endif

__device__ __forceinline__ unsigned lds_addr_of(const void* p) {
  return (unsigned)(size_t)p;  // flat LDS aperture keeps offset in addr[31:0]
}

// ---------------------------------------------------------------- convert
__global__ void f32_to_bf16(const float* __restrict__ in,
                            unsigned short* __restrict__ out, int n) {
  int i = blockIdx.x * blockDim.x + threadIdx.x;
  int stride = gridDim.x * blockDim.x;
  for (; i < n; i += stride) out[i] = f2bf(in[i]);
}

// ---------------------------------------------------------------- GEMM
// C[m][n] = alpha * (sum_k A[m][k] * B[n][k] + bias[n])
// A: MxK bf16 row-major, B: NxK bf16 row-major (i.e. computes A @ B^T)
// 256 threads = 8 waves; block tile 128x128, BK = 32 (one WMMA k-step).
// A/B tiles staged by TDM (double buffered); wave 0 drives the DMA.
template <bool OUT_F32>
__global__ __launch_bounds__(256) void gemm_bf16t(
    const unsigned short* __restrict__ A, const unsigned short* __restrict__ B,
    const float* __restrict__ bias, unsigned short* __restrict__ Cb,
    float* __restrict__ Cf, int M, int N, int K, float alpha) {
  __shared__ __align__(16) unsigned short As[2][128][32];
  __shared__ __align__(16) unsigned short Bs[2][128][32];
  const int tid = threadIdx.x;
  const int lane = tid & 31, wv = tid >> 5;
  const int lane16 = lane & 15, h2 = lane >> 4;   // C-layout: row = v + 8*h2
  const int wrow = wv >> 2, wcol = wv & 3;        // wave -> 64x32 region
  const int m0 = blockIdx.y * 128, n0 = blockIdx.x * 128;

  v8f zero = {0.f, 0.f, 0.f, 0.f, 0.f, 0.f, 0.f, 0.f};
  v8f acc[4][2];
#pragma unroll
  for (int a = 0; a < 4; ++a)
#pragma unroll
    for (int b2 = 0; b2 < 2; ++b2) acc[a][b2] = zero;

#if HAVE_TDM
  const unsigned asb = lds_addr_of(&As[0][0][0]);
  const unsigned bsb = lds_addr_of(&Bs[0][0][0]);
  if (wv == 0) {  // prologue: stage k-step 0 into buffer 0
    tdm_load_2d(asb, A + (size_t)m0 * K, 32, 128, K);
    tdm_load_2d(bsb, B + (size_t)n0 * K, 32, 128, K);
  }
#endif

  int buf = 0;
  for (int kt = 0; kt < K; kt += 32) {
#if HAVE_TDM
    if (wv == 0) {
      __builtin_amdgcn_s_wait_tensorcnt(0);       // current tiles landed
      if (kt + 32 < K) {                          // prefetch next k-step
        tdm_load_2d(asb + (unsigned)(buf ^ 1) * 8192u,
                    A + (size_t)m0 * K + kt + 32, 32, 128, K);
        tdm_load_2d(bsb + (unsigned)(buf ^ 1) * 8192u,
                    B + (size_t)n0 * K + kt + 32, 32, 128, K);
      }
    }
    __syncthreads();
#else
#pragma unroll
    for (int i = 0; i < 2; ++i) {                 // 512 16B chunks / 256 thr
      int c = tid * 2 + i;
      int r = c >> 2, off = (c & 3) * 8;
      *(u16x8*)&As[buf][r][off] =
          *(const u16x8*)&A[(size_t)(m0 + r) * K + kt + off];
      *(u16x8*)&Bs[buf][r][off] =
          *(const u16x8*)&B[(size_t)(n0 + r) * K + kt + off];
    }
    __syncthreads();
#endif

    FragU bfr[2];
#pragma unroll
    for (int nt = 0; nt < 2; ++nt) {              // B: 8-elem K chunks interleaved
      int nl = wcol * 32 + nt * 16 + lane16;
      bfr[nt].u[0] = *(const u16x8*)&Bs[buf][nl][h2 * 8];
      bfr[nt].u[1] = *(const u16x8*)&Bs[buf][nl][16 + h2 * 8];
    }
#pragma unroll
    for (int mt = 0; mt < 4; ++mt) {              // A: 8-elem K chunks interleaved
      int ml = wrow * 64 + mt * 16 + lane16;
      FragU afr;
      afr.u[0] = *(const u16x8*)&As[buf][ml][h2 * 8];
      afr.u[1] = *(const u16x8*)&As[buf][ml][16 + h2 * 8];
#pragma unroll
      for (int nt = 0; nt < 2; ++nt) {
        acc[mt][nt] = __builtin_amdgcn_wmma_f32_16x16x32_bf16(
            false, afr.bf, false, bfr[nt].bf, (short)0, acc[mt][nt], false,
            false);
      }
    }
    __syncthreads();
    buf ^= 1;
  }

#pragma unroll
  for (int nt = 0; nt < 2; ++nt) {
    int n = n0 + wcol * 32 + nt * 16 + lane16;
    float bv = bias[n];
#pragma unroll
    for (int mt = 0; mt < 4; ++mt) {
#pragma unroll
      for (int v = 0; v < 8; ++v) {
        int m = m0 + wrow * 64 + mt * 16 + v + 8 * h2;
        float val = alpha * (acc[mt][nt][v] + bv);
        if (OUT_F32)
          Cf[(size_t)m * N + n] = val;
        else
          Cb[(size_t)m * N + n] = f2bf(val);
      }
    }
  }
}

// ---------------------------------------------------------------- attention
// Flash-attention: grid = (seq/64, H, bs), 128 threads = 4 waves.
// Wave w owns 16 query rows; streams key/value blocks of 32.
// K tile staged by TDM; V loaded cooperatively (needs transpose for B layout).
__global__ __launch_bounds__(128) void attn_kernel(
    const unsigned short* __restrict__ Q, const unsigned short* __restrict__ Kg,
    const unsigned short* __restrict__ Vg, unsigned short* __restrict__ O,
    int seq, int H) {
  const int hidden = 1024;  // H * 64
  __shared__ __align__(16) unsigned short Ks[32][64];      // [key][d]
  __shared__ __align__(16) unsigned short Vt[64][32];      // [d][key] (transposed)
  __shared__ __align__(16) unsigned short Ps[4][16][32];   // per-wave P scratch

  const int tid = threadIdx.x;
  const int lane = tid & 31, wv = tid >> 5;
  const int lane16 = lane & 15, h2 = lane >> 4;
  const int b = blockIdx.z, h = blockIdx.y, qblk = blockIdx.x;
  const int qr0 = qblk * 64 + wv * 16;
  const size_t rowbase = (size_t)b * seq;
#if HAVE_TDM
  const unsigned ksb = lds_addr_of(&Ks[0][0]);
#endif

  // Q fragments for the two d-halves (scale 1/8 already folded into Q).
  FragU qf[2];
  {
    const unsigned short* qp = Q + (rowbase + qr0 + lane16) * hidden + h * 64;
#pragma unroll
    for (int kk = 0; kk < 2; ++kk) {
      qf[kk].u[0] = *(const u16x8*)(qp + kk * 32 + h2 * 8);
      qf[kk].u[1] = *(const u16x8*)(qp + kk * 32 + 16 + h2 * 8);
    }
  }

  v8f zero = {0.f, 0.f, 0.f, 0.f, 0.f, 0.f, 0.f, 0.f};
  v8f o[4];
  float mrow[8], lrow[8];
#pragma unroll
  for (int t = 0; t < 4; ++t) o[t] = zero;
#pragma unroll
  for (int v = 0; v < 8; ++v) { mrow[v] = -3.0e38f; lrow[v] = 0.f; }

  const int nblk = qblk * 2 + 2;  // causal: keys up to qblk*64+63
  for (int j = 0; j < nblk; ++j) {
#if HAVE_TDM
    if (wv == 0)   // K tile: 32 rows x 64 elems, row stride = hidden
      tdm_load_2d(ksb, Kg + (rowbase + j * 32) * hidden + h * 64, 64, 32,
                  (unsigned)hidden);
#endif
#pragma unroll
    for (int i = 0; i < 2; ++i) {  // cooperative V load (transposed)
      int c = tid * 2 + i;
      int r = c >> 3, off = (c & 7) * 8;
      size_t g = (rowbase + j * 32 + r) * hidden + h * 64 + off;
#if !HAVE_TDM
      *(u16x8*)&Ks[r][off] = *(const u16x8*)&Kg[g];
#endif
      u16x8 vvec = *(const u16x8*)&Vg[g];
#pragma unroll
      for (int e = 0; e < 8; ++e) Vt[off + e][r] = vvec[e];
    }
#if HAVE_TDM
    if (wv == 0) __builtin_amdgcn_s_wait_tensorcnt(0);
#endif
    __syncthreads();

    // S = Q @ K^T  (16 x 32, two 16x16 C tiles)
    v8f s[2];
#pragma unroll
    for (int nt = 0; nt < 2; ++nt) {
      int n = nt * 16 + lane16;
      v8f accs = zero;
#pragma unroll
      for (int kk = 0; kk < 2; ++kk) {
        FragU bf;
        bf.u[0] = *(const u16x8*)&Ks[n][kk * 32 + h2 * 8];
        bf.u[1] = *(const u16x8*)&Ks[n][kk * 32 + 16 + h2 * 8];
        accs = __builtin_amdgcn_wmma_f32_16x16x32_bf16(
            false, qf[kk].bf, false, bf.bf, (short)0, accs, false, false);
      }
      s[nt] = accs;
    }

    // Online softmax (rows live in VGPR index + lane-half; cols across lanes)
#pragma unroll
    for (int v = 0; v < 8; ++v) {
      int qrow = qr0 + v + 8 * h2;
      float s0 = s[0][v], s1 = s[1][v];
      if (j * 32 + lane16 > qrow) s0 = -3.0e38f;
      if (j * 32 + 16 + lane16 > qrow) s1 = -3.0e38f;
      float mt = fmaxf(s0, s1);
#pragma unroll
      for (int off = 8; off; off >>= 1)
        mt = fmaxf(mt, __shfl_xor(mt, off, 16));
      float mnew = fmaxf(mrow[v], mt);
      float aold = __expf(mrow[v] - mnew);
      float e0 = __expf(s0 - mnew);
      float e1 = __expf(s1 - mnew);
      float rs = e0 + e1;
#pragma unroll
      for (int off = 8; off; off >>= 1) rs += __shfl_xor(rs, off, 16);
      lrow[v] = lrow[v] * aold + rs;
      mrow[v] = mnew;
#pragma unroll
      for (int t = 0; t < 4; ++t) o[t][v] *= aold;
      Ps[wv][v + 8 * h2][lane16] = f2bf(e0);
      Ps[wv][v + 8 * h2][16 + lane16] = f2bf(e1);
    }

    // P in A-fragment layout via per-wave LDS round trip
    FragU pf;
    pf.u[0] = *(const u16x8*)&Ps[wv][lane16][h2 * 8];
    pf.u[1] = *(const u16x8*)&Ps[wv][lane16][16 + h2 * 8];

    // O += P @ V   (V transposed in LDS -> interleaved 8-elem B fragments)
#pragma unroll
    for (int dt = 0; dt < 4; ++dt) {
      FragU bf;
      int dc = dt * 16 + lane16;
      bf.u[0] = *(const u16x8*)&Vt[dc][h2 * 8];
      bf.u[1] = *(const u16x8*)&Vt[dc][16 + h2 * 8];
      o[dt] = __builtin_amdgcn_wmma_f32_16x16x32_bf16(
          false, pf.bf, false, bf.bf, (short)0, o[dt], false, false);
    }
    __syncthreads();
  }

#pragma unroll
  for (int v = 0; v < 8; ++v) {
    float inv = 1.0f / lrow[v];
    int row = qr0 + v + 8 * h2;
    size_t base = (rowbase + row) * hidden + h * 64;
#pragma unroll
    for (int dt = 0; dt < 4; ++dt)
      O[base + dt * 16 + lane16] = f2bf(o[dt][v] * inv);
  }
}

// ---------------------------------------------------------------- launcher
extern "C" void kernel_launch(void* const* d_in, const int* in_sizes, int n_in,
                              void* d_out, int out_size, void* d_ws,
                              size_t ws_size, hipStream_t stream) {
  (void)in_sizes; (void)n_in; (void)out_size; (void)ws_size;
  const float* x  = (const float*)d_in[0];
  const float* Wq = (const float*)d_in[1];
  const float* bq = (const float*)d_in[2];
  const float* Wk = (const float*)d_in[3];
  const float* bk = (const float*)d_in[4];
  const float* Wv = (const float*)d_in[5];
  const float* bv = (const float*)d_in[6];
  const float* Wm = (const float*)d_in[7];
  const float* bm = (const float*)d_in[8];
  float* out = (float*)d_out;

  const int bs = 2, seq = 2048, hidden = 1024, K = 1024, H = 16;
  const int M = bs * seq;                     // 4096
  const size_t NM = (size_t)M * hidden;       // 4 Mi elems
  const size_t NW = (size_t)hidden * K;       // 1 Mi elems

  unsigned short* w = (unsigned short*)d_ws;
  unsigned short* xb  = w;  w += NM;
  unsigned short* wqb = w;  w += NW;
  unsigned short* wkb = w;  w += NW;
  unsigned short* wvb = w;  w += NW;
  unsigned short* wmb = w;  w += NW;
  unsigned short* Qb  = w;  w += NM;
  unsigned short* Kb  = w;  w += NM;
  unsigned short* Vb  = w;  w += NM;
  unsigned short* Ab  = w;  w += NM;

  f32_to_bf16<<<512, 256, 0, stream>>>(x, xb, (int)NM);
  f32_to_bf16<<<256, 256, 0, stream>>>(Wq, wqb, (int)NW);
  f32_to_bf16<<<256, 256, 0, stream>>>(Wk, wkb, (int)NW);
  f32_to_bf16<<<256, 256, 0, stream>>>(Wv, wvb, (int)NW);
  f32_to_bf16<<<256, 256, 0, stream>>>(Wm, wmb, (int)NW);

  dim3 gg(hidden / 128, M / 128);
  // scale 1/sqrt(64) folded into Q projection
  gemm_bf16t<false><<<gg, 256, 0, stream>>>(xb, wqb, bq, Qb, nullptr, M, hidden, K, 0.125f);
  gemm_bf16t<false><<<gg, 256, 0, stream>>>(xb, wkb, bk, Kb, nullptr, M, hidden, K, 1.0f);
  gemm_bf16t<false><<<gg, 256, 0, stream>>>(xb, wvb, bv, Vb, nullptr, M, hidden, K, 1.0f);

  attn_kernel<<<dim3(seq / 64, H, bs), 128, 0, stream>>>(Qb, Kb, Vb, Ab, seq, H);

  gemm_bf16t<true><<<gg, 256, 0, stream>>>(Ab, wmb, bm, nullptr, out, M, hidden, K, 1.0f);
}
